// LSTMNet_74586402062649
// MI455X (gfx1250) — compile-verified, matrix operand order checked by reading, not run on hardware
//
#include <hip/hip_runtime.h>

// ---------------------------------------------------------------------------
// LSTM on MI455X (gfx1250): bf16 WMMA GEMMs with f32 accumulation.
//   Phase 0: convert x -> bf16; pack W,U into WMMA-B-fragment layout (bf16).
//   Phase 1: xW = x @ W + bias  (big parallel GEMM, v_wmma_f32_16x16x32_bf16)
//   Phase 2: 512 sequential step kernels: gates = xW[s] + h @ U, fused gate
//            math on the f32 accumulators, double-buffered bf16 h.
//   Phase 3: emit (h_T, c_T).
// Software-pipelined GEMM loops: fragments for k-tile kt+1 are loaded while
// the 4 WMMAs of k-tile kt execute, so loads overlap matrix ops.
// ---------------------------------------------------------------------------

typedef __bf16 bf16_t;
typedef __attribute__((ext_vector_type(16))) __bf16 v16bf;
typedef __attribute__((ext_vector_type(8)))  __bf16 v8bf;
typedef __attribute__((ext_vector_type(8)))  float  v8f;

#define BATCH 64
#define SEQ   512
#define DIN   1024
#define HID   1024
#define GATES 4096          // 4*HID
#define KT32  32            // K=1024 in chunks of 32

// A-fragment (16x32 bf16, ISA 7.12.2): lane L holds row m=L%16;
// element e<8 -> k = (L/16)*8 + e ; e>=8 -> k = 16 + (L/16)*8 + (e-8).
// Both halves are contiguous 8-element (16B) runs -> two b128 loads.
__device__ __forceinline__ v16bf load_a_frag(const bf16_t* row, int k0) {
    v8bf a0 = *(const v8bf*)(row + k0);        // k0 = kt*32 + (lane>>4)*8
    v8bf a1 = *(const v8bf*)(row + k0 + 16);
    v16bf a;
#pragma unroll
    for (int i = 0; i < 8; ++i) { a[i] = a0[i]; a[i + 8] = a1[i]; }
    return a;
}

__device__ __forceinline__ float sigmoidf_dev(float x) {
    return 1.0f / (1.0f + __expf(-x));
}

// --------------------------- Phase 0 kernels -------------------------------

__global__ void cvt_x_bf16(const float* __restrict__ x, bf16_t* __restrict__ xb, int n) {
    int i = blockIdx.x * blockDim.x + threadIdx.x;
    if (i < n) xb[i] = (bf16_t)x[i];
}

// Pack a (1024 x 4096) f32 matrix into WMMA-B-fragment bf16 layout:
// dst[((nt*32 + kt)*32 + lane)*16 + e]  with lane: n=lane%16, kbase=(lane/16)*8
// so each lane's 16 bf16 values for one (kt,nt) tile are one 32B chunk.
__global__ void pack_b_frag(const float* __restrict__ src, bf16_t* __restrict__ dst) {
    int tid  = blockIdx.x * blockDim.x + threadIdx.x;   // 256*32*32*16 threads
    int e    = tid & 15;
    int lane = (tid >> 4) & 31;
    int kt   = (tid >> 9) & 31;
    int nt   = tid >> 14;
    int kb   = (lane >> 4) * 8;
    int k    = kt * 32 + ((e < 8) ? (kb + e) : (16 + kb + (e - 8)));
    int n    = nt * 16 + (lane & 15);
    dst[tid] = (bf16_t)src[(size_t)k * GATES + n];
}

__global__ void init_state(float* __restrict__ c, bf16_t* __restrict__ h0,
                           bf16_t* __restrict__ h1) {
    int i = blockIdx.x * blockDim.x + threadIdx.x;      // 64*1024 threads
    c[i]  = 0.0f;
    h0[i] = (bf16_t)0.0f;
    h1[i] = (bf16_t)0.0f;
}

// --------------------------- Phase 1: xW GEMM ------------------------------
// Rows r = s*64+b (32768), cols = 4096. grid = (512, 64), block = 128 (4 waves).
// Wave w: m-tile mt = 4*blockIdx.x + w; 4 n-tiles ntb..ntb+3 (A reused 4x).
__global__ void gemm_xw(const bf16_t* __restrict__ xb, const bf16_t* __restrict__ Wp,
                        const float* __restrict__ bias, float* __restrict__ xW) {
    int wave = threadIdx.x >> 5;
    int lane = threadIdx.x & 31;
    int mt   = blockIdx.x * 4 + wave;
    int ntb  = blockIdx.y * 4;
    int half = lane >> 4;

    // A row for this lane: global row r = mt*16 + (lane%16); x is (B,S,D)
    int r = mt * 16 + (lane & 15);
    int srow = r >> 6, brow = r & 63;
    const bf16_t* arow = xb + ((size_t)brow * SEQ + srow) * DIN;
    const bf16_t* bbase = Wp + ((size_t)ntb * KT32 * 32 + lane) * 16;

    // ---- software pipeline: preload k-tile 0 ----
    v8f acc[4] = {};
    v16bf a = load_a_frag(arow, half * 8);
    v16bf bfr[4];
#pragma unroll
    for (int g = 0; g < 4; ++g)
        bfr[g] = *(const v16bf*)(bbase + (size_t)g * (KT32 * 32 * 16));

    for (int kt = 0; kt < KT32; ++kt) {
        v16bf a_n;
        v16bf b_n[4];
        if (kt + 1 < KT32) {                 // issue next tile's loads first
            int k1 = (kt + 1) * 32 + half * 8;
            __builtin_prefetch(arow + k1 + 128, 0, 3);   // global_prefetch_b8, near scope
            a_n = load_a_frag(arow, k1);
#pragma unroll
            for (int g = 0; g < 4; ++g)
                b_n[g] = *(const v16bf*)(bbase + (size_t)g * (KT32 * 32 * 16)
                                               + (size_t)(kt + 1) * (32 * 16));
        }
#pragma unroll
        for (int g = 0; g < 4; ++g)
            acc[g] = __builtin_amdgcn_wmma_f32_16x16x32_bf16(
                false, a, false, bfr[g], (short)0, acc[g], false, false);
        if (kt + 1 < KT32) {
            a = a_n;
#pragma unroll
            for (int g = 0; g < 4; ++g) bfr[g] = b_n[g];
        }
    }

    // C layout: VGPR v -> row m = v + 8*(lane/16), col n = lane%16
#pragma unroll
    for (int g = 0; g < 4; ++g) {
        int n  = (ntb + g) * 16 + (lane & 15);
        float bv = bias[n];
#pragma unroll
        for (int v = 0; v < 8; ++v) {
            int r2 = mt * 16 + v + 8 * half;
            xW[(size_t)r2 * GATES + n] = acc[g][v] + bv;
        }
    }
}

// --------------------------- Phase 2: LSTM step ----------------------------
// grid = 64 (hidden col tile ht), block = 128 (4 waves = 4 batch row tiles).
// Each wave computes the i/f/g/o 16x16 tiles for the SAME hidden columns
// (gate n-tiles ht, ht+64, ht+128, ht+192) so gate math fuses in-register.
__global__ void lstm_step(int s,
                          const float*  __restrict__ xW,
                          const bf16_t* __restrict__ Up,
                          const bf16_t* __restrict__ h_in,
                          bf16_t*       __restrict__ h_out,
                          float*        __restrict__ cst,
                          float*        __restrict__ hidden) {
    int wave = threadIdx.x >> 5;
    int lane = threadIdx.x & 31;
    int ht   = blockIdx.x;          // 0..63
    int mt   = wave;                // 0..3 -> batch rows
    int half = lane >> 4;
    int col  = ht * 16 + (lane & 15);           // hidden column

    const bf16_t* arow  = h_in + (size_t)(mt * 16 + (lane & 15)) * HID;
    const bf16_t* bbase = Up + ((size_t)ht * KT32 * 32 + lane) * 16;

    // Warm the tail data while the GEMM runs: xW gate rows + c state.
#pragma unroll
    for (int v = 0; v < 8; ++v) {
        int b = mt * 16 + v + 8 * half;
        __builtin_prefetch(xW + ((size_t)s * BATCH + b) * GATES + col, 0, 3);
        __builtin_prefetch(cst + b * HID + col, 0, 3);
    }

    // ---- software pipeline: preload k-tile 0 ----
    v8f acc[4] = {};
    v16bf a = load_a_frag(arow, half * 8);
    v16bf bfr[4];
#pragma unroll
    for (int g = 0; g < 4; ++g)     // gate g n-tile = ht + 64*g
        bfr[g] = *(const v16bf*)(bbase + (size_t)g * (64 * KT32 * 32 * 16));

    for (int kt = 0; kt < KT32; ++kt) {
        v16bf a_n;
        v16bf b_n[4];
        if (kt + 1 < KT32) {
            int k1 = (kt + 1) * 32 + half * 8;
            a_n = load_a_frag(arow, k1);
#pragma unroll
            for (int g = 0; g < 4; ++g)
                b_n[g] = *(const v16bf*)(bbase + (size_t)g * (64 * KT32 * 32 * 16)
                                               + (size_t)(kt + 1) * (32 * 16));
        }
#pragma unroll
        for (int g = 0; g < 4; ++g)
            acc[g] = __builtin_amdgcn_wmma_f32_16x16x32_bf16(
                false, a, false, bfr[g], (short)0, acc[g], false, false);
        if (kt + 1 < KT32) {
            a = a_n;
#pragma unroll
            for (int g = 0; g < 4; ++g) bfr[g] = b_n[g];
        }
    }

#pragma unroll
    for (int v = 0; v < 8; ++v) {
        int b = mt * 16 + v + 8 * half;         // batch row
        size_t xwb = ((size_t)s * BATCH + b) * GATES + col;
        float gi = acc[0][v] + xW[xwb];
        float gf = acc[1][v] + xW[xwb + HID];
        float gg = acc[2][v] + xW[xwb + 2 * HID];
        float go = acc[3][v] + xW[xwb + 3 * HID];

        float it = sigmoidf_dev(gi);
        float ft = sigmoidf_dev(gf);
        float gt = tanhf(gg);
        float ot = sigmoidf_dev(go);

        int ci = b * HID + col;
        float cn = ft * cst[ci] + it * gt;
        cst[ci]  = cn;
        float h  = ot * tanhf(cn);

        hidden[((size_t)b * SEQ + s) * HID + col] = h;
        h_out[ci] = (bf16_t)h;
    }
}

// --------------------------- Phase 3: finalize -----------------------------

__global__ void finalize(const float* __restrict__ hidden, const float* __restrict__ cst,
                         float* __restrict__ hT, float* __restrict__ cT) {
    int i = blockIdx.x * blockDim.x + threadIdx.x;      // 64*1024 threads
    int b = i >> 10, hc = i & 1023;
    hT[i] = hidden[((size_t)b * SEQ + (SEQ - 1)) * HID + hc];
    cT[i] = cst[i];
}

// ---------------------------------------------------------------------------

extern "C" void kernel_launch(void* const* d_in, const int* in_sizes, int n_in,
                              void* d_out, int out_size, void* d_ws, size_t ws_size,
                              hipStream_t stream) {
    const float* x    = (const float*)d_in[0];   // (64,512,1024)
    const float* W    = (const float*)d_in[1];   // (1024,4096)
    const float* U    = (const float*)d_in[2];   // (1024,4096)
    const float* bias = (const float*)d_in[3];   // (4096,)

    float* out_hidden = (float*)d_out;                         // (64,512,1024)
    float* out_hT     = out_hidden + (size_t)BATCH * SEQ * HID;
    float* out_cT     = out_hT + (size_t)BATCH * HID;

    // Workspace layout (offsets keep >=256B alignment)
    char* ws = (char*)d_ws;
    float*  xW  = (float*)ws;                                          // 512 MB
    size_t  off = (size_t)SEQ * BATCH * GATES * sizeof(float);
    bf16_t* xb  = (bf16_t*)(ws + off);  off += (size_t)BATCH * SEQ * DIN * sizeof(bf16_t);   // 64 MB
    bf16_t* Wp  = (bf16_t*)(ws + off);  off += (size_t)DIN * GATES * sizeof(bf16_t);         // 8 MB
    bf16_t* Up  = (bf16_t*)(ws + off);  off += (size_t)HID * GATES * sizeof(bf16_t);         // 8 MB
    float*  cst = (float*) (ws + off);  off += (size_t)BATCH * HID * sizeof(float);
    bf16_t* h0  = (bf16_t*)(ws + off);  off += (size_t)BATCH * HID * sizeof(bf16_t);
    bf16_t* h1  = (bf16_t*)(ws + off);

    // Phase 0
    cvt_x_bf16<<<(BATCH * SEQ * DIN) / 256, 256, 0, stream>>>(x, xb, BATCH * SEQ * DIN);
    pack_b_frag<<<(256 * 32 * 32 * 16) / 256, 256, 0, stream>>>(W, Wp);
    pack_b_frag<<<(256 * 32 * 32 * 16) / 256, 256, 0, stream>>>(U, Up);
    init_state<<<(BATCH * HID) / 256, 256, 0, stream>>>(cst, h0, h1);

    // Phase 1: xW = x @ W + bias   (32768 x 4096, K=1024)
    gemm_xw<<<dim3(512, 64), 128, 0, stream>>>(xb, Wp, bias, xW);

    // Phase 2: sequential recurrence, stream order = inter-step barrier
    bf16_t* hin = h0;
    bf16_t* hout = h1;
    for (int s = 0; s < SEQ; ++s) {
        lstm_step<<<64, 128, 0, stream>>>(s, xW, Up, hin, hout, cst, out_hidden);
        bf16_t* t = hin; hin = hout; hout = t;
    }

    // Phase 3
    finalize<<<(BATCH * HID) / 256, 256, 0, stream>>>(out_hidden, cst, out_hT, out_cT);
}
